// GCN3_Lin1_56581899158201
// MI455X (gfx1250) — matrix operand order repaired and checked by reading, not compiled, not run on hardware
//
#include <hip/hip_runtime.h>
#include <hip/hip_bf16.h>

typedef __attribute__((ext_vector_type(16))) _Float16 v16h;
typedef __attribute__((ext_vector_type(8)))  float    v8f;

// ---------------------------------------------------------------------------
// Graph normalization (computed once, reused for all 3 layers)
// ---------------------------------------------------------------------------
__global__ void zero_f32(float* __restrict__ p, int n) {
  int i = blockIdx.x * blockDim.x + threadIdx.x;
  if (i < n) p[i] = 0.0f;
}

__global__ void deg_scatter(const int* __restrict__ dst, float* __restrict__ deg, int E) {
  int e = blockIdx.x * blockDim.x + threadIdx.x;
  if (e < E) atomicAdd(&deg[dst[e]], 1.0f);
}

// deg -> (deg+1)^-1/2 in place (+1 = self-loop, so always > 0)
__global__ void dinv_kernel(float* __restrict__ deg, int n) {
  int i = blockIdx.x * blockDim.x + threadIdx.x;
  if (i < n) deg[i] = rsqrtf(deg[i] + 1.0f);
}

// norm[e] = dinv[src]*dinv[dst] for edges; norm[E+i] = dinv[i]^2 for self-loops
__global__ void norm_kernel(const int* __restrict__ src, const int* __restrict__ dst,
                            const float* __restrict__ dinv, float* __restrict__ norm,
                            int E, int n) {
  int t = blockIdx.x * blockDim.x + threadIdx.x;
  if (t < E) {
    norm[t] = dinv[src[t]] * dinv[dst[t]];
  } else if (t < E + n) {
    float d = dinv[t - E];
    norm[t] = d * d;
  }
}

// ---------------------------------------------------------------------------
// Pack W[kin,nout] into transposed, zero-padded Wp[32 cols][32 k]:
//   Wp[col*32 + k] = W[k*nout + col]   (0 outside)
// ---------------------------------------------------------------------------
__global__ void pack_w(const float* __restrict__ W, float* __restrict__ Wp,
                       int kin, int nout) {
  int t = blockIdx.x * blockDim.x + threadIdx.x;
  if (t < 1024) {
    int col = t >> 5, k = t & 31;
    Wp[t] = (col < nout && k < kin) ? W[k * nout + col] : 0.0f;
  }
}

// ---------------------------------------------------------------------------
// Tiny GEMM  H[n,NOUT] = X[n,KIN] @ W  via v_wmma_f32_16x16x32_f16.
// One wave per 16x16 output tile. KIN/NOUT compile-time => branchless fragment
// loads (A: 2x b128 contiguous; B: 4x b128 from padded Wp) and unconditional
// stores (H rows padded to a multiple of 16; only the column guard survives,
// hoisted to a single branch, and it is compile-time true for NOUT=32).
// ---------------------------------------------------------------------------
template<int KIN, int NOUT>
__global__ void wmma_gemm_t(const float* __restrict__ X, const float* __restrict__ Wp,
                            float* __restrict__ H, int n) {
  static_assert(KIN == 16 || KIN == 24 || KIN == 32, "");
  constexpr int NT = (NOUT + 15) / 16;
  int gwave = (blockIdx.x * blockDim.x + threadIdx.x) >> 5;
  int lane  = threadIdx.x & 31;
  int mt = gwave / NT;
  int nt = gwave % NT;
  int m0 = mt << 4;
  int n0 = nt << 4;
  if (m0 >= n) return;                  // wave-uniform exit; EXEC full at WMMA

  int  hl = lane & 15;
  bool hi = lane >= 16;
  int  row  = m0 + hl;
  int  rowc = row < n ? row : (n - 1);  // branchless clamp (tail tile only)
  int  col  = n0 + hl;                  // Wp is padded; always safe

  // ---- A fragment (16-bit A 16x32 layout) ----
  // element i (0..7):  K = base + i          (base = hi ? 8 : 0)
  // element i (8..15): K = 16 + base + (i-8)
  v16h a;
  {
    const float* xr = X + (size_t)rowc * KIN;
    int base = hi ? 8 : 0;
    float4 p0 = *(const float4*)(xr + base);
    float4 p1 = *(const float4*)(xr + base + 4);
    a[0] = (_Float16)p0.x; a[1] = (_Float16)p0.y;
    a[2] = (_Float16)p0.z; a[3] = (_Float16)p0.w;
    a[4] = (_Float16)p1.x; a[5] = (_Float16)p1.y;
    a[6] = (_Float16)p1.z; a[7] = (_Float16)p1.w;
    if constexpr (KIN == 32) {
      float4 q0 = *(const float4*)(xr + 16 + base);
      float4 q1 = *(const float4*)(xr + 20 + base);
      a[8]  = (_Float16)q0.x; a[9]  = (_Float16)q0.y;
      a[10] = (_Float16)q0.z; a[11] = (_Float16)q0.w;
      a[12] = (_Float16)q1.x; a[13] = (_Float16)q1.y;
      a[14] = (_Float16)q1.z; a[15] = (_Float16)q1.w;
    } else if constexpr (KIN == 24) {
      // K=16..23 valid only for lanes 0-15; hi lanes zeroed branchlessly.
      // (Both halves load K=16..23, which lie inside the 24-float row.)
      float4 q0 = *(const float4*)(xr + 16);
      float4 q1 = *(const float4*)(xr + 20);
      float z = hi ? 0.0f : 1.0f;
      a[8]  = (_Float16)(q0.x * z); a[9]  = (_Float16)(q0.y * z);
      a[10] = (_Float16)(q0.z * z); a[11] = (_Float16)(q0.w * z);
      a[12] = (_Float16)(q1.x * z); a[13] = (_Float16)(q1.y * z);
      a[14] = (_Float16)(q1.z * z); a[15] = (_Float16)(q1.w * z);
    } else { // KIN == 16: K >= 16 is zero padding
      #pragma unroll
      for (int i = 8; i < 16; ++i) a[i] = (_Float16)0.0f;
    }
  }

  // ---- B fragment: lane = column, element i: K = (hi?8:0) + i (i<8), 16 + (hi?8:0) + (i-8) (i>=8) ----
  v16h b;
  {
    const float* wc = Wp + (size_t)col * 32 + (hi ? 8 : 0);
    float4 w0 = *(const float4*)(wc + 0);
    float4 w1 = *(const float4*)(wc + 4);
    float4 w2 = *(const float4*)(wc + 16);
    float4 w3 = *(const float4*)(wc + 20);
    b[0]  = (_Float16)w0.x; b[1]  = (_Float16)w0.y;
    b[2]  = (_Float16)w0.z; b[3]  = (_Float16)w0.w;
    b[4]  = (_Float16)w1.x; b[5]  = (_Float16)w1.y;
    b[6]  = (_Float16)w1.z; b[7]  = (_Float16)w1.w;
    b[8]  = (_Float16)w2.x; b[9]  = (_Float16)w2.y;
    b[10] = (_Float16)w2.z; b[11] = (_Float16)w2.w;
    b[12] = (_Float16)w3.x; b[13] = (_Float16)w3.y;
    b[14] = (_Float16)w3.z; b[15] = (_Float16)w3.w;
  }

  v8f c = {};
  c = __builtin_amdgcn_wmma_f32_16x16x32_f16(
      /*neg_a=*/false, a, /*neg_b=*/false, b,
      /*c_mod=*/(short)0, c, /*reuse_a=*/false, /*reuse_b=*/false);

  // C layout: VGPR r <-> row (m0 + r + 8*hi), col = lane&15.
  // H is row-padded, so no row guard; single hoisted column guard.
  float* hp = H + (size_t)(m0 + (hi ? 8 : 0)) * NOUT + col;
  bool cok = (NOUT % 16 == 0) ? true : (col < NOUT);
  if (cok) {
    #pragma unroll
    for (int r = 0; r < 8; ++r) hp[(size_t)r * NOUT] = c[r];
  }
}

// ---------------------------------------------------------------------------
// Aggregation: agg initialized with the self-loop message (doubles as the
// zero-fill), then edges scatter-add, then bias+ReLU in place.
// ---------------------------------------------------------------------------
template<int DIM>
__global__ void init_agg(const float* __restrict__ h, const float* __restrict__ selfnorm,
                         float* __restrict__ agg, int n) {
  int t = blockIdx.x * blockDim.x + threadIdx.x;
  if (t < n * DIM) agg[t] = h[t] * selfnorm[t / DIM];
}

__global__ void edge_scatter(const int* __restrict__ src, const int* __restrict__ dst,
                             const float* __restrict__ norm, const float* __restrict__ h,
                             float* __restrict__ agg, int E, int dim) {
  int e = blockIdx.x * blockDim.x + threadIdx.x;
  if (e >= E) return;
  int   s = src[e];
  int   d = dst[e];
  float w = norm[e];
  const float4* hs = (const float4*)(h + (size_t)s * dim);
  float* ad = agg + (size_t)d * dim;
  int q = dim >> 2;                     // dim 32/24/12 -> 8/6/3 float4 chunks
  for (int f4 = 0; f4 < q; ++f4) {
    float4 v = hs[f4];
    atomicAdd(ad + f4 * 4 + 0, v.x * w);
    atomicAdd(ad + f4 * 4 + 1, v.y * w);
    atomicAdd(ad + f4 * 4 + 2, v.z * w);
    atomicAdd(ad + f4 * 4 + 3, v.w * w);
  }
}

template<int DIM>
__global__ void bias_relu(float* __restrict__ agg, const float* __restrict__ b, int n) {
  int t = blockIdx.x * blockDim.x + threadIdx.x;
  if (t < n * DIM) {
    float v = agg[t] + b[t % DIM];
    agg[t] = v > 0.0f ? v : 0.0f;
  }
}

// ---------------------------------------------------------------------------
// Head: logits = X[n,12] @ Wf[12,6] + bf; log_softmax over 6 classes.
// ---------------------------------------------------------------------------
__global__ void final_head(const float* __restrict__ X, const float* __restrict__ Wf,
                           const float* __restrict__ bf, float* __restrict__ out, int n) {
  int i = blockIdx.x * blockDim.x + threadIdx.x;
  if (i >= n) return;
  float l[6];
  #pragma unroll
  for (int j = 0; j < 6; ++j) l[j] = bf[j];
  #pragma unroll
  for (int k = 0; k < 12; ++k) {
    float xv = X[i * 12 + k];
    #pragma unroll
    for (int j = 0; j < 6; ++j) l[j] = fmaf(xv, Wf[k * 6 + j], l[j]);
  }
  float m = l[0];
  #pragma unroll
  for (int j = 1; j < 6; ++j) m = fmaxf(m, l[j]);
  float s = 0.0f;
  #pragma unroll
  for (int j = 0; j < 6; ++j) s += expf(l[j] - m);
  float lse = logf(s) + m;
  #pragma unroll
  for (int j = 0; j < 6; ++j) out[i * 6 + j] = l[j] - lse;
}

// ---------------------------------------------------------------------------
static inline int cdiv(int a, int b) { return (a + b - 1) / b; }

template<int KIN, int NOUT>
static void run_gcn_layer(const float* X, const float* W, const float* b, float* Wp,
                          const int* src, const int* dst, const float* norm,
                          const float* selfnorm, float* h, float* agg,
                          int N, int E, hipStream_t stream) {
  pack_w<<<4, 256, 0, stream>>>(W, Wp, KIN, NOUT);
  int waves = cdiv(N, 16) * cdiv(NOUT, 16);
  wmma_gemm_t<KIN, NOUT><<<cdiv(waves * 32, 256), 256, 0, stream>>>(X, Wp, h, N);
  init_agg<NOUT><<<cdiv(N * NOUT, 256), 256, 0, stream>>>(h, selfnorm, agg, N);
  edge_scatter<<<cdiv(E, 256), 256, 0, stream>>>(src, dst, norm, h, agg, E, NOUT);
  bias_relu<NOUT><<<cdiv(N * NOUT, 256), 256, 0, stream>>>(agg, b, N);
}

extern "C" void kernel_launch(void* const* d_in, const int* in_sizes, int n_in,
                              void* d_out, int out_size, void* d_ws, size_t ws_size,
                              hipStream_t stream) {
  const float* x  = (const float*)d_in[0];
  const int*   ei = (const int*)d_in[1];
  const float* W1 = (const float*)d_in[2];
  const float* b1 = (const float*)d_in[3];
  const float* W2 = (const float*)d_in[4];
  const float* b2 = (const float*)d_in[5];
  const float* W3 = (const float*)d_in[6];
  const float* b3 = (const float*)d_in[7];
  const float* Wf = (const float*)d_in[8];
  const float* bf = (const float*)d_in[9];
  float* out = (float*)d_out;

  const int N = in_sizes[0] / 16;   // 100000
  const int E = in_sizes[1] / 2;    // 6400000
  const int* src = ei;              // edge_index[0]
  const int* dst = ei + E;          // edge_index[1]

  // Workspace layout (floats). Feature buffers get 16 rows of padding so the
  // WMMA store path needs no row guards (tail-tile writes land in padding).
  const size_t FCAP = (size_t)(N + 16) * 32;
  float* ws = (float*)d_ws;
  size_t off = 0;
  float* Wp   = ws + off; off += 1024;                       // packed weights
  float* dinv = ws + off; off += (size_t)cdiv(N, 64) * 64;
  float* norm = ws + off; off += (size_t)cdiv(E + N, 64) * 64;
  float* h    = ws + off; off += FCAP;
  float* bufA = ws + off; off += FCAP;
  float* bufB = ws + off; off += FCAP;
  const float* selfnorm = norm + E;

  // Graph normalization (once, reused by all layers)
  zero_f32<<<cdiv(N, 256), 256, 0, stream>>>(dinv, N);
  deg_scatter<<<cdiv(E, 256), 256, 0, stream>>>(dst, dinv, E);
  dinv_kernel<<<cdiv(N, 256), 256, 0, stream>>>(dinv, N);
  norm_kernel<<<cdiv(E + N, 256), 256, 0, stream>>>(src, dst, dinv, norm, E, N);

  // Three GCN layers (agg buffer becomes next layer's input)
  run_gcn_layer<16, 32>(x,    W1, b1, Wp, src, dst, norm, selfnorm, h, bufA, N, E, stream);
  run_gcn_layer<32, 24>(bufA, W2, b2, Wp, src, dst, norm, selfnorm, h, bufB, N, E, stream);
  run_gcn_layer<24, 12>(bufB, W3, b3, Wp, src, dst, norm, selfnorm, h, bufA, N, E, stream);

  // Final linear + log_softmax
  final_head<<<cdiv(N, 256), 256, 0, stream>>>(bufA, Wf, bf, out, N);

  (void)n_in; (void)out_size; (void)ws_size;
}
